// MultiHeadedAttention_88794153877525
// MI455X (gfx1250) — matrix operand order repaired and checked by reading, not compiled, not run on hardware
//
#include <hip/hip_runtime.h>
#include <hip/hip_bf16.h>

typedef __bf16 bf16;
typedef __attribute__((ext_vector_type(16))) __bf16 v16bf;
typedef __attribute__((ext_vector_type(2)))  __bf16 bf16x2;
typedef __attribute__((ext_vector_type(8)))  float   v8f;

#define WMMA_BF16(a, b, c) \
    __builtin_amdgcn_wmma_f32_16x16x32_bf16(false, (a), false, (b), (short)0, (c), false, false)

// ---------------------------------------------------------------------------
// Fragment loaders.  Layouts per CDNA5 ISA 7.12.2 (wave32):
//   A (16x32, 16-bit): lane l<16 -> row l, VGPR v: K = (v&4?16:0) + 2*(v&3)
//                      lane l>=16 -> row l-16, +8 on K
//   B (32x16, 16-bit): lane: N = lane&15; VGPR v: K = (v&4?16:0) + 2*(v&3), +8 for lane>=16
//   C/D (16x16 f32):   VGPR r: lanes 0-15 -> M=r, lanes 16-31 -> M=r+8; N=lane&15
// Pair (2 x bf16) elements are contiguous in memory -> 32-bit LDS loads that
// the compiler coalesces into ds_load_b128.
// ---------------------------------------------------------------------------
__device__ __forceinline__ v16bf load_frag_a(const bf16* p, int stride, int lane, int koff) {
    v16bf a;
    const int row = lane & 15;
    const int hi  = (lane >> 4) & 1;
    const bf16* base = p + row * stride + koff;
#pragma unroll
    for (int v = 0; v < 8; ++v) {
        const int k = ((v & 4) ? 16 : 0) + hi * 8 + (v & 3) * 2;
        bf16x2 pr = *(const bf16x2*)(base + k);
        a[2 * v]     = pr[0];
        a[2 * v + 1] = pr[1];
    }
    return a;
}

__device__ __forceinline__ v16bf load_frag_b(const bf16* p, int stride, int lane, int koff) {
    v16bf b;
    const int n  = lane & 15;
    const int hi = (lane >> 4) & 1;
    const bf16* base = p + n * stride + koff + hi * 8;
#pragma unroll
    for (int v = 0; v < 8; ++v) {
        bf16x2 pr = *(const bf16x2*)(base + ((v & 4) ? 16 : 0) + (v & 3) * 2);
        b[2 * v]     = pr[0];
        b[2 * v + 1] = pr[1];
    }
    return b;
}

// ---------------------------------------------------------------------------
// Kernel 1: fused Q/K/V projection.  Y = X @ W^T  (W stored [out,in]).
// Grid: (D/64, BS/64, 3). Block: 128 (4 waves). Each wave: 16x64 f32 tile.
// Q, K stored [B][H][S][64] bf16; V stored TRANSPOSED [B][H][64][S] bf16 so the
// attention kernel can async-copy V^T tiles straight into B-fragment layout.
// Q is pre-scaled by 1/sqrt(d_k) = 0.125.
// ---------------------------------------------------------------------------
__global__ __launch_bounds__(128)
void mha_qkv_proj(const float* __restrict__ X,
                  const float* __restrict__ Wq, const float* __restrict__ Wk,
                  const float* __restrict__ Wv,
                  bf16* __restrict__ Qw, bf16* __restrict__ Kw, bf16* __restrict__ Vw) {
    constexpr int LS = 40;  // LDS row stride (32 + pad, even)
    __shared__ __align__(16) bf16 Xs[64 * LS];
    __shared__ __align__(16) bf16 Ws[64 * LS];

    const int tid  = threadIdx.x;
    const int lane = tid & 31;
    const int wave = tid >> 5;
    const int rowbase = blockIdx.y * 64;   // [0, 8192)
    const int colbase = blockIdx.x * 64;   // [0, 1024)
    const float* W  = (blockIdx.z == 0) ? Wq : (blockIdx.z == 1) ? Wk : Wv;
    bf16*       dst = (blockIdx.z == 0) ? Qw : (blockIdx.z == 1) ? Kw : Vw;
    const float scale = (blockIdx.z == 0) ? 0.125f : 1.0f;
    const bool  vsel  = (blockIdx.z == 2);

    v8f acc[4];
#pragma unroll
    for (int t = 0; t < 4; ++t)
#pragma unroll
        for (int r = 0; r < 8; ++r) acc[t][r] = 0.0f;

    for (int k0 = 0; k0 < 1024; k0 += 32) {
        __syncthreads();
#pragma unroll
        for (int i = 0; i < 16; ++i) {
            const int e = tid + i * 128;
            const int r = e >> 5, c = e & 31;
            Xs[r * LS + c] = (bf16)X[(size_t)(rowbase + r) * 1024 + k0 + c];
            Ws[r * LS + c] = (bf16)W[(size_t)(colbase + r) * 1024 + k0 + c];
        }
        __syncthreads();

        const v16bf a = load_frag_a(Xs + wave * 16 * LS, LS, lane, 0);
#pragma unroll
        for (int t = 0; t < 4; ++t) {
            const v16bf b = load_frag_b(Ws + t * 16 * LS, LS, lane, 0);
            acc[t] = WMMA_BF16(a, b, acc[t]);
        }
    }

    const int hi = (lane >> 4) & 1, nl = lane & 15;
#pragma unroll
    for (int t = 0; t < 4; ++t)
#pragma unroll
        for (int r = 0; r < 8; ++r) {
            const int m  = r + hi * 8;
            const int sg = rowbase + wave * 16 + m;   // global token index
            const int o  = colbase + t * 16 + nl;     // output feature
            const int b  = sg >> 11, s = sg & 2047;
            const int h  = o >> 6,  dk = o & 63;
            const size_t head = (size_t)(b * 16 + h);
            const size_t idx = vsel ? (head * 64 + dk) * 2048 + s     // V^T: [B][H][dk][S]
                                    : (head * 2048 + s) * 64 + dk;    // Q,K: [B][H][S][dk]
            dst[idx] = (bf16)(acc[t][r] * scale);
        }
}

// ---------------------------------------------------------------------------
// Kernel 2: flash attention per (b,h).  Grid: (S/64, B*H). Block: 128.
// K/V^T tiles are staged global->LDS with the CDNA5 async copy engine
// (GLOBAL_LOAD_ASYNC_TO_LDS_B128, ASYNCcnt), double-buffered so staging of
// block kb+1 overlaps WMMA compute on block kb.
// ---------------------------------------------------------------------------
#define ATTN_LS      80                      // LDS row stride in elements (160 B, 16B-aligned)
#define ATTN_TILE_B  (64 * ATTN_LS * 2)      // one 64-row tile: 10240 bytes
#define ATTN_LDS_B   (5 * ATTN_TILE_B)       // 2xK + 2xV + P regions = 51200 bytes

__device__ __forceinline__ void async_copy_row_b128(unsigned lds_byte_off,
                                                    const bf16* gbase,
                                                    unsigned g_byte_off) {
    // GVS mode: mem_addr = SADDR(64b) + VADDR(32b) ; 16 bytes/lane -> LDS.
    asm volatile("global_load_async_to_lds_b128 %0, %1, %2 offset:0"
                 :
                 : "v"(lds_byte_off), "v"(g_byte_off), "s"(gbase)
                 : "memory");
}

__global__ __launch_bounds__(128)
void mha_flash_attn(const bf16* __restrict__ Qw, const bf16* __restrict__ Kw,
                    const bf16* __restrict__ Vw, bf16* __restrict__ Hw) {
    extern __shared__ __align__(16) bf16 smem[];
    constexpr int LS = ATTN_LS;

    const int tid  = threadIdx.x;
    const int lane = tid & 31;
    const int wave = tid >> 5;
    const int bh    = blockIdx.y;        // 0..63
    const int qbase = blockIdx.x * 64;   // query block base within S
    const bf16* Qg = Qw + (size_t)bh * 2048 * 64;   // [S][64]
    const bf16* Kg = Kw + (size_t)bh * 2048 * 64;   // [S][64]
    const bf16* Vg = Vw + (size_t)bh * 64 * 2048;   // V^T: [64][S]

    // LDS regions (byte offsets relative to dynamic-LDS base)
    const unsigned ldsbase = __builtin_amdgcn_groupstaticsize();  // dyn LDS starts here (0)
    bf16* Kbuf = smem;                    // 2 x [64][LS]  (key-major)
    bf16* Vbuf = smem + 2 * 64 * LS;      // 2 x [64][LS]  (dk-major, key-contiguous)
    bf16* Ps   = smem + 4 * 64 * LS;      // 4 waves x [16][LS]
    const unsigned kls0 = ldsbase;
    const unsigned vls0 = ldsbase + 2u * ATTN_TILE_B;

    // Per-thread async staging pattern: 64 rows x 8 chunks of 16B = 512 chunks,
    // 4 chunks per thread per tile.
    const int srow = tid >> 3;            // base row   (0..15), +16 per j
    const int sch  = tid & 7;             // 16B chunk within row (0..7)

    // Q fragments for this wave's 16 rows (K = d_k = 64 -> two 32-wide A frags)
    const bf16* qrow = Qg + (size_t)(qbase + wave * 16) * 64;
    const v16bf qf0 = load_frag_a(qrow, 64, lane, 0);
    const v16bf qf1 = load_frag_a(qrow, 64, lane, 32);

    v8f oacc[4];
    float rm[8], rl[8];
#pragma unroll
    for (int r = 0; r < 8; ++r) { rm[r] = -3.0e38f; rl[r] = 0.0f; }
#pragma unroll
    for (int t = 0; t < 4; ++t)
#pragma unroll
        for (int r = 0; r < 8; ++r) oacc[t][r] = 0.0f;

    bf16* Pw = Ps + wave * 16 * LS;

    // ---- issue async stage of key-block 0 into buffer 0 ----
#pragma unroll
    for (int j = 0; j < 4; ++j) {
        const int row = srow + j * 16;
        async_copy_row_b128(kls0 + (unsigned)(row * (LS * 2) + sch * 16),
                            Kg, (unsigned)((row * 64 + sch * 8) * 2));
        async_copy_row_b128(vls0 + (unsigned)(row * (LS * 2) + sch * 16),
                            Vg, (unsigned)((row * 2048 + sch * 8) * 2));
    }

    for (int kb = 0; kb < 32; ++kb) {
        const int buf = kb & 1;
        __syncthreads();   // all waves done computing on buffer buf^1 (block kb-1)

        if (kb + 1 < 32) {   // stage next block into the other buffer (overlaps compute)
            const int nbuf = buf ^ 1;
#pragma unroll
            for (int j = 0; j < 4; ++j) {
                const int row = srow + j * 16;
                async_copy_row_b128(kls0 + (unsigned)(nbuf * ATTN_TILE_B + row * (LS * 2) + sch * 16),
                                    Kg, (unsigned)((((kb + 1) * 64 + row) * 64 + sch * 8) * 2));
                async_copy_row_b128(vls0 + (unsigned)(nbuf * ATTN_TILE_B + row * (LS * 2) + sch * 16),
                                    Vg, (unsigned)((row * 2048 + (kb + 1) * 64 + sch * 8) * 2));
            }
            // 8 newly issued remain outstanding; wait for the 8 of block kb
            asm volatile("s_wait_asynccnt 8" ::: "memory");
        } else {
            asm volatile("s_wait_asynccnt 0" ::: "memory");
        }
        __syncthreads();   // block kb resident in LDS for every wave

        const bf16* Ksb = Kbuf + buf * 64 * LS;
        const bf16* Vsb = Vbuf + buf * 64 * LS;

        // ---- scores: S = Q x K^T (16x64 per wave), f32 accum ----
        v8f sacc[4];
#pragma unroll
        for (int t = 0; t < 4; ++t) {
#pragma unroll
            for (int r = 0; r < 8; ++r) sacc[t][r] = 0.0f;
            const v16bf b0 = load_frag_b(Ksb + t * 16 * LS, LS, lane, 0);
            sacc[t] = WMMA_BF16(qf0, b0, sacc[t]);
            const v16bf b1 = load_frag_b(Ksb + t * 16 * LS, LS, lane, 32);
            sacc[t] = WMMA_BF16(qf1, b1, sacc[t]);
        }

        // ---- online softmax on C-layout (row r / r+8 per half-wave) ----
#pragma unroll
        for (int r = 0; r < 8; ++r) {
            float mx = fmaxf(fmaxf(sacc[0][r], sacc[1][r]),
                             fmaxf(sacc[2][r], sacc[3][r]));
#pragma unroll
            for (int off = 8; off >= 1; off >>= 1)
                mx = fmaxf(mx, __shfl_xor(mx, off, 32));
            const float mnew  = fmaxf(rm[r], mx);
            const float alpha = __expf(rm[r] - mnew);
            rm[r] = mnew;
            float psum = 0.0f;
#pragma unroll
            for (int t = 0; t < 4; ++t) {
                const float p = __expf(sacc[t][r] - mnew);
                psum += p;
                sacc[t][r] = p;
            }
#pragma unroll
            for (int off = 8; off >= 1; off >>= 1)
                psum += __shfl_xor(psum, off, 32);
            rl[r] = rl[r] * alpha + psum;
#pragma unroll
            for (int t = 0; t < 4; ++t) oacc[t][r] *= alpha;
        }

        // ---- spill P (C-layout) to per-wave LDS, re-read as A fragments ----
        {
            const int hi = (lane >> 4) & 1, nl = lane & 15;
#pragma unroll
            for (int r = 0; r < 8; ++r) {
                const int m = r + hi * 8;
#pragma unroll
                for (int t = 0; t < 4; ++t)
                    Pw[m * LS + t * 16 + nl] = (bf16)sacc[t][r];
            }
        }
        // per-wave LDS RAW: wait for our ds stores before re-reading
        asm volatile("s_wait_dscnt 0" ::: "memory");

        // ---- O += P x V  (V^T tile already in B-fragment orientation) ----
#pragma unroll
        for (int ks = 0; ks < 2; ++ks) {
            const v16bf ap = load_frag_a(Pw, LS, lane, ks * 32);
#pragma unroll
            for (int t = 0; t < 4; ++t) {
                const v16bf bv = load_frag_b(Vsb + t * 16 * LS, LS, lane, ks * 32);
                oacc[t] = WMMA_BF16(ap, bv, oacc[t]);
            }
        }
    }

    // epilogue: normalize and write H_cat as bf16 [B][S][1024]
    const int b = bh >> 4, h = bh & 15;
    const int hi = (lane >> 4) & 1, nl = lane & 15;
#pragma unroll
    for (int r = 0; r < 8; ++r) {
        const float inv = 1.0f / rl[r];
        const int m = r + hi * 8;
        const int s = qbase + wave * 16 + m;
#pragma unroll
        for (int t = 0; t < 4; ++t)
            Hw[((size_t)b * 2048 + s) * 1024 + h * 64 + t * 16 + nl] =
                (bf16)(oacc[t][r] * inv);
    }
}

// ---------------------------------------------------------------------------
// Kernel 3: output projection.  out = H_cat @ W_h^T + b_h  (f32 result).
// Grid: (D/64, BS/64). Block: 128.
// ---------------------------------------------------------------------------
__global__ __launch_bounds__(128)
void mha_out_proj(const bf16* __restrict__ H, const float* __restrict__ Wh,
                  const float* __restrict__ bias, float* __restrict__ out) {
    constexpr int LS = 40;
    __shared__ __align__(16) bf16 Xs[64 * LS];
    __shared__ __align__(16) bf16 Ws[64 * LS];

    const int tid  = threadIdx.x;
    const int lane = tid & 31;
    const int wave = tid >> 5;
    const int rowbase = blockIdx.y * 64;
    const int colbase = blockIdx.x * 64;

    v8f acc[4];
#pragma unroll
    for (int t = 0; t < 4; ++t)
#pragma unroll
        for (int r = 0; r < 8; ++r) acc[t][r] = 0.0f;

    for (int k0 = 0; k0 < 1024; k0 += 32) {
        __syncthreads();
#pragma unroll
        for (int i = 0; i < 16; ++i) {
            const int e = tid + i * 128;
            const int r = e >> 5, c = e & 31;
            Xs[r * LS + c] = H[(size_t)(rowbase + r) * 1024 + k0 + c];
            Ws[r * LS + c] = (bf16)Wh[(size_t)(colbase + r) * 1024 + k0 + c];
        }
        __syncthreads();

        const v16bf a = load_frag_a(Xs + wave * 16 * LS, LS, lane, 0);
#pragma unroll
        for (int t = 0; t < 4; ++t) {
            const v16bf b = load_frag_b(Ws + t * 16 * LS, LS, lane, 0);
            acc[t] = WMMA_BF16(a, b, acc[t]);
        }
    }

    const int hi = (lane >> 4) & 1, nl = lane & 15;
#pragma unroll
    for (int t = 0; t < 4; ++t)
#pragma unroll
        for (int r = 0; r < 8; ++r) {
            const int m  = r + hi * 8;
            const int sg = rowbase + wave * 16 + m;
            const int o  = colbase + t * 16 + nl;
            out[(size_t)sg * 1024 + o] = acc[t][r] + bias[o];
        }
}

// ---------------------------------------------------------------------------
extern "C" void kernel_launch(void* const* d_in, const int* in_sizes, int n_in,
                              void* d_out, int out_size, void* d_ws, size_t ws_size,
                              hipStream_t stream) {
    const float* X  = (const float*)d_in[0];
    const float* Wq = (const float*)d_in[1];
    const float* Wk = (const float*)d_in[2];
    const float* Wv = (const float*)d_in[3];
    const float* Wh = (const float*)d_in[4];
    const float* bh = (const float*)d_in[5];
    float* out = (float*)d_out;

    // workspace: 4 bf16 tensors of 8,388,608 elements (16 MB each)
    const size_t T = (size_t)4 * 16 * 2048 * 64;
    bf16* ws = (bf16*)d_ws;
    bf16* Qw = ws;            // [B][H][S][64]
    bf16* Kw = ws + T;        // [B][H][S][64]
    bf16* Vw = ws + 2 * T;    // [B][H][64][S]  (transposed)
    bf16* Hw = ws + 3 * T;    // [B][S][1024]

    const dim3 blk(128);
    mha_qkv_proj  <<<dim3(16, 128, 3), blk, 0,          stream>>>(X, Wq, Wk, Wv, Qw, Kw, Vw);
    mha_flash_attn<<<dim3(32, 64),     blk, ATTN_LDS_B, stream>>>(Qw, Kw, Vw, Hw);
    mha_out_proj  <<<dim3(16, 128),    blk, 0,          stream>>>(Hw, Wh, bh, out);
}
